// Model_21921513079169
// MI455X (gfx1250) — matrix operand order repaired and checked by reading, not compiled, not run on hardware
//
#include <hip/hip_runtime.h>
#include <hip/hip_bf16.h>

// ---------------------------------------------------------------------------
// Types for CDNA5 WMMA
// ---------------------------------------------------------------------------
typedef __attribute__((ext_vector_type(16))) _Float16 v16h;
typedef __attribute__((ext_vector_type(8)))  _Float16 v8h;
typedef __attribute__((ext_vector_type(8)))  float    v8f;

#define BM 64
#define BN 32
#define BK 32

#define FLAG_RELU   1
#define FLAG_ADDRES 2

// ---------------------------------------------------------------------------
// Double-buffered WMMA tile loop.  FULL=true: no bounds handling on X
// (interior n-block, K multiple of BK).  FULL=false: min-clamped addresses,
// mask-multiply zero-fill (keeps loads unconditional -> no load sinking).
// ---------------------------------------------------------------------------
template<bool FULL>
static __device__ __forceinline__ v8f run_tiles(
    const _Float16* __restrict__ Wh, const float* __restrict__ Xb,
    _Float16 (*As)[BM][BK + 8], _Float16 (*XsT)[BN][BK + 8],
    int M, int N, int K, int ldx, int m0, int n0, int tid)
{
    const int lane = tid & 31;
    const int wid  = tid >> 5;
    const int wm   = wid & 3;            // 4 wave tiles along M
    const int wn   = wid >> 2;           // 2 wave tiles along N
    const int r    = lane & 15;
    const int hf   = lane >> 4;
    const int mmA  = tid >> 2;           // 0..63
    const int kbA  = (tid & 3) << 3;     // 0,8,16,24

    v8h   wreg;
    float xreg[4];

    auto loadA = [&](int k0) {
        const int gm = m0 + mmA;
        const int gk = k0 + kbA;
        const bool ok = (gm < M) && (gk < K);          // 8-chunks fully in/out
        const size_t idx = (size_t)min(gm, M - 1) * K + min(gk, K - 8);
        v8h t = *(const v8h*)(Wh + idx);               // unconditional b128 load
        const v8h zz = {};
        wreg = ok ? t : zz;
    };
    auto loadX = [&](int k0) {
#pragma unroll
        for (int i = 0; i < 4; ++i) {
            int e  = tid + i * 256;
            int kk = e >> 5;
            int nn = e & (BN - 1);
            int gk = k0 + kk, gn = n0 + nn;
            if (FULL) {
                xreg[i] = Xb[(size_t)gk * ldx + gn];   // fully unguarded
            } else {
                float xv  = Xb[(size_t)min(gk, K - 1) * ldx + min(gn, N - 1)];
                float msk = ((gk < K) && (gn < N)) ? 1.0f : 0.0f;
                xreg[i] = xv * msk;    // value always used -> load not sinkable
            }
        }
    };

    v8f acc = {};
    loadA(0);
    loadX(0);

    int buf = 0;
    for (int k0 = 0; k0 < K; k0 += BK) {
        // ---- commit staged registers to LDS[buf] ----
        *(v8h*)&As[buf][mmA][kbA] = wreg;              // ds_store_b128
#pragma unroll
        for (int i = 0; i < 4; ++i) {
            int e = tid + i * 256;
            XsT[buf][e & (BN - 1)][e >> 5] = (_Float16)xreg[i];
        }
        __syncthreads();

        // ---- prefetch next tile while this one computes ----
        if (k0 + BK < K) {
            loadA(k0 + BK);
            loadX(k0 + BK);
        }

        // A fragment: lanes 0-15 row M=r hold K {0..7,16..23}; lanes 16-31 hold {8..15,24..31}
        const v8h a0 = *(const v8h*)&As [buf][wm * 16 + r][hf * 8];
        const v8h a1 = *(const v8h*)&As [buf][wm * 16 + r][16 + hf * 8];
        // B fragment: lanes 0-15 col N=r hold K {0..7,16..23}; lanes 16-31 hold {8..15,24..31}
        const v8h b0 = *(const v8h*)&XsT[buf][wn * 16 + r][hf * 8];
        const v8h b1 = *(const v8h*)&XsT[buf][wn * 16 + r][16 + hf * 8];
        v16h af = __builtin_shufflevector(a0, a1, 0,1,2,3,4,5,6,7,8,9,10,11,12,13,14,15);
        v16h bf = __builtin_shufflevector(b0, b1, 0,1,2,3,4,5,6,7,8,9,10,11,12,13,14,15);

        acc = __builtin_amdgcn_wmma_f32_16x16x32_f16(
            /*neg_a=*/false, af, /*neg_b=*/false, bf,
            /*c_mod=*/(short)0, acc, /*reuse_a=*/false, /*reuse_b=*/false);

        buf ^= 1;
    }
    return acc;
}

// ---------------------------------------------------------------------------
// Fused GEMM + BN/bias (+ residual) (+ ReLU), batched over blockIdx.z.
//   out[b, m, n] = epi( sum_k W[m,k] * X[b, k, n] )
//   W is pre-converted f16 (M x K, K multiple of 8, K >= 24).
//   X row stride = ldx, X batch stride = K*ldx.  out/res strides = (M*N, N).
// ---------------------------------------------------------------------------
__global__ __launch_bounds__(256) void gemm_bn_kernel(
    const _Float16* __restrict__ Wh, const float* __restrict__ X,
    const float* __restrict__ scale, const float* __restrict__ shift,
    const float* __restrict__ res, float* __restrict__ out,
    int M, int N, int K, int ldx, int flags)
{
    __shared__ __align__(16) _Float16 As [2][BM][BK + 8];   // W tiles
    __shared__ __align__(16) _Float16 XsT[2][BN][BK + 8];   // X tiles, (n,k)

    const int b    = blockIdx.z;
    const int m0   = blockIdx.y * BM;
    const int n0   = blockIdx.x * BN;
    const int tid  = threadIdx.x;
    const int lane = tid & 31;
    const int wid  = tid >> 5;
    const int wm   = wid & 3;
    const int wn   = wid >> 2;
    const int r    = lane & 15;
    const int hf   = lane >> 4;

    const float* __restrict__ Xb = X + (size_t)b * (size_t)K * (size_t)ldx;

    // Block-uniform fast-path selection (scalar branch, no divergence).
    const bool full = (n0 + BN <= N) && ((K & (BK - 1)) == 0);

    v8f acc;
    if (full) acc = run_tiles<true >(Wh, Xb, As, XsT, M, N, K, ldx, m0, n0, tid);
    else      acc = run_tiles<false>(Wh, Xb, As, XsT, M, N, K, ldx, m0, n0, tid);

    // Epilogue: D VGPR i -> M = i + 8*hf, N = r
    const int gn = n0 + wn * 16 + r;
    if (gn < N) {
#pragma unroll
        for (int i = 0; i < 8; ++i) {
            int gm = m0 + wm * 16 + i + 8 * hf;
            if (gm < M) {
                float v = acc[i] * scale[gm] + shift[gm];
                size_t off = ((size_t)b * M + gm) * (size_t)N + gn;
                if (flags & FLAG_ADDRES) v += res[off];
                if (flags & FLAG_RELU)   v = fmaxf(v, 0.0f);
                out[off] = v;
            }
        }
    }
}

// ---------------------------------------------------------------------------
// f32 -> f16 weight conversion (run once per launch into ws pool)
// ---------------------------------------------------------------------------
__global__ void f32_to_f16_kernel(const float* __restrict__ in,
                                  _Float16* __restrict__ out, int n)
{
    int i = blockIdx.x * blockDim.x + threadIdx.x;
    if (i < n) out[i] = (_Float16)in[i];
}

// ---------------------------------------------------------------------------
// Precontract Gw with EW / NS:  GE[f,j] = sum_d Gw[f,j,d]*EW[f,d]
// ---------------------------------------------------------------------------
__global__ void prep_ge_kernel(float* __restrict__ GE, float* __restrict__ GN,
                               const float* __restrict__ Gw,
                               const float* __restrict__ EW,
                               const float* __restrict__ NS, int nf)
{
    int i = blockIdx.x * blockDim.x + threadIdx.x;   // f*3 + j
    if (i >= nf * 3) return;
    int f = i / 3;
    float ge = 0.f, gn = 0.f;
#pragma unroll
    for (int d = 0; d < 3; ++d) {
        float g = Gw[(size_t)i * 3 + d];
        ge += g * EW[(size_t)f * 3 + d];
        gn += g * NS[(size_t)f * 3 + d];
    }
    GE[i] = ge;
    GN[i] = gn;
}

// ---------------------------------------------------------------------------
// Fold conv bias + BN into per-channel scale/shift:
//   bn(conv+bias) = conv*s + (bias-m)*s + b,  s = g/sqrt(v+eps)
// ---------------------------------------------------------------------------
__global__ void fold_bn_kernel(float* __restrict__ scale, float* __restrict__ shift,
                               const float* __restrict__ bias,
                               const float* __restrict__ g, const float* __restrict__ b,
                               const float* __restrict__ m, const float* __restrict__ v, int C)
{
    int c = blockIdx.x * blockDim.x + threadIdx.x;
    if (c >= C) return;
    float s = g[c] * rsqrtf(v[c] + 1e-5f);
    scale[c] = s;
    shift[c] = (bias[c] - m[c]) * s + b[c];
}

// ---------------------------------------------------------------------------
// Fused mesh operator: builds feat (B, C*4, nvp) = [identity, lap, vew, vns]
// F[(b*C+c)*4 + k][v].  GE/GN precontracted; 26 gathers per thread.
// ---------------------------------------------------------------------------
__global__ void build_feat_kernel(const float* __restrict__ X, float* __restrict__ F,
                                  const int* __restrict__ L_idx, const float* __restrict__ L_w,
                                  const int* __restrict__ Fv_idx,
                                  const float* __restrict__ GE, const float* __restrict__ GN,
                                  const int* __restrict__ F2V_idx, const float* __restrict__ F2V_w,
                                  int total, int Vin, int nvp)
{
    int idx = blockIdx.x * blockDim.x + threadIdx.x;
    if (idx >= total) return;
    int v  = idx % nvp;
    int bc = idx / nvp;                 // b*C + c
    const float* __restrict__ Xr = X + (size_t)bc * Vin;

    float idv = Xr[v];
    float lap = 0.f;
#pragma unroll
    for (int t = 0; t < 7; ++t)
        lap += Xr[L_idx[v * 7 + t]] * L_w[v * 7 + t];

    float vew = 0.f, vns = 0.f;
#pragma unroll
    for (int k = 0; k < 6; ++k) {
        int   f = F2V_idx[v * 6 + k];
        float w = F2V_w  [v * 6 + k];
        float ew = 0.f, ns = 0.f;
#pragma unroll
        for (int j = 0; j < 3; ++j) {
            float xv = Xr[Fv_idx[f * 3 + j]];
            ew += xv * GE[f * 3 + j];
            ns += xv * GN[f * 3 + j];
        }
        vew += w * ew;
        vns += w * ns;
    }
    float* Fp = F + ((size_t)bc * 4) * nvp + v;
    Fp[0]               = idv;
    Fp[(size_t)nvp]     = lap;
    Fp[(size_t)2 * nvp] = vew;
    Fp[(size_t)3 * nvp] = vns;
}

// ---------------------------------------------------------------------------
// Mean over vertex dim: pooled[b*C+c] = mean_v X[b,c,v]
// ---------------------------------------------------------------------------
__global__ void mean_pool_kernel(const float* __restrict__ X, float* __restrict__ out,
                                 int BC, int V)
{
    int i = blockIdx.x * blockDim.x + threadIdx.x;
    if (i >= BC) return;
    const float* p = X + (size_t)i * V;
    float s = 0.f;
    for (int v = 0; v < V; ++v) s += p[v];
    out[i] = s / (float)V;
}

// ---------------------------------------------------------------------------
// FC(2048 -> 40) + log_softmax.  One block per batch row.
// ---------------------------------------------------------------------------
__global__ void fc_lsm_kernel(const float* __restrict__ pooled,
                              const float* __restrict__ Wfc,
                              const float* __restrict__ bfc,
                              float* __restrict__ out)
{
    __shared__ float logits[40];
    __shared__ float red[2];
    const int b = blockIdx.x;
    const int t = threadIdx.x;    // 64 threads
    if (t < 40) {
        const float* p = pooled + (size_t)b * 2048;
        const float* w = Wfc + (size_t)t * 2048;
        float s = bfc[t];
        for (int k = 0; k < 2048; ++k) s += p[k] * w[k];
        logits[t] = s;
    }
    __syncthreads();
    if (t == 0) {
        float mx = -3.4e38f;
        for (int o = 0; o < 40; ++o) mx = fmaxf(mx, logits[o]);
        float se = 0.f;
        for (int o = 0; o < 40; ++o) se += expf(logits[o] - mx);
        red[0] = mx;
        red[1] = logf(se);
    }
    __syncthreads();
    if (t < 40) out[(size_t)b * 40 + t] = logits[t] - red[0] - red[1];
}

// ---------------------------------------------------------------------------
// Host side
// ---------------------------------------------------------------------------
static inline int ceil_div(int a, int b) { return (a + b - 1) / b; }

// Input indices (depth-first flattening of setup_inputs dicts)
#define IN_X        0
#define IN_COEFFS   1
#define IN_BIAS     2
#define IN_BN       3   // g,b,m,v = 3..6
#define B1_BASE     7   // w1,b1,coeffs,cbias,w3,b3,ws,bs, bn1(4), bn2(4), bn3(4), bns(4)
#define B2_BASE     31
#define B3_BASE     55
#define IN_FCW      79
#define IN_FCB      80
#define M5_BASE     81  // L_idx,L_w,Fv_idx,Gw,EW,NS,F2V_idx,F2V_w
#define M3_BASE     89
#define M2_BASE     97
#define M1_BASE     105

extern "C" void kernel_launch(void* const* d_in, const int* in_sizes, int n_in,
                              void* d_out, int out_size, void* d_ws, size_t ws_size,
                              hipStream_t stream)
{
    (void)in_sizes; (void)n_in; (void)out_size; (void)ws_size;

    const int B = 64;
    // mesh sizes
    const int NV5 = 10242, NVP5 = 2562, NF5 = 20480;
    const int NV3 = 642,  NF3 = 1280;
    const int NV2 = 162,  NF2 = 320;
    const int NV1 = 42,   NF1 = 80;

    float* w = (float*)d_ws;
    size_t off = 0;
    auto alloc = [&](size_t n) { float* p = w + off; off += n; return p; };

    float* GE5 = alloc((size_t)NF5 * 3); float* GN5 = alloc((size_t)NF5 * 3);
    float* GE3 = alloc((size_t)NF3 * 3); float* GN3 = alloc((size_t)NF3 * 3);
    float* GE2 = alloc((size_t)NF2 * 3); float* GN2 = alloc((size_t)NF2 * 3);
    float* GE1 = alloc((size_t)NF1 * 3); float* GN1 = alloc((size_t)NF1 * 3);

    // folded BN scale/shift per epilogue
    float *sc_in = alloc(32),  *sh_in = alloc(32);
    float *sc11 = alloc(32),   *sh11 = alloc(32);
    float *sc12 = alloc(32),   *sh12 = alloc(32);
    float *sc13 = alloc(128),  *sh13 = alloc(128);
    float *sc1s = alloc(128),  *sh1s = alloc(128);
    float *sc21 = alloc(128),  *sh21 = alloc(128);
    float *sc22 = alloc(128),  *sh22 = alloc(128);
    float *sc23 = alloc(512),  *sh23 = alloc(512);
    float *sc2s = alloc(512),  *sh2s = alloc(512);
    float *sc31 = alloc(512),  *sh31 = alloc(512);
    float *sc32 = alloc(512),  *sh32 = alloc(512);
    float *sc33 = alloc(2048), *sh33 = alloc(2048);
    float *sc3s = alloc(2048), *sh3s = alloc(2048);

    const size_t ACT = 5505024;     // max of 64*{32*2562, 128*642, 512*162, 2048*42, feat}
    float* P0 = alloc(ACT);
    float* P1 = alloc(ACT);
    float* F  = alloc(ACT);         // feat buffer; reused as residual (h3) buffer
    float* H1 = alloc(1376256);
    float* H2 = alloc(1376256);
    float* pooled = alloc((size_t)B * 2048);

    // ---- f16 weight pool (16B-aligned; every K multiple of 8 keeps rows aligned)
    off = (off + 3) & ~(size_t)3;
    _Float16* hp = (_Float16*)(w + off);
    size_t hoff = 0;
    auto halloc = [&](size_t n) { _Float16* p = hp + hoff;
                                  hoff += (n + 7) & ~(size_t)7; return p; };

    _Float16* Wc0 = halloc(32 * 24);
    _Float16* W11 = halloc(32 * 32);
    _Float16* W12 = halloc(32 * 128);
    _Float16* W13 = halloc(128 * 32);
    _Float16* W1S = halloc(128 * 32);
    _Float16* W21 = halloc(128 * 128);
    _Float16* W22 = halloc(128 * 512);
    _Float16* W23 = halloc(512 * 128);
    _Float16* W2S = halloc(512 * 128);
    _Float16* W31 = halloc(512 * 512);
    _Float16* W32 = halloc(512 * 2048);
    _Float16* W33 = halloc(2048 * 512);
    _Float16* W3S = halloc(2048 * 512);

    // -------- helpers --------
    auto cvt = [&](int wi, _Float16* dst, int n) {
        f32_to_f16_kernel<<<ceil_div(n, 256), 256, 0, stream>>>(
            (const float*)d_in[wi], dst, n);
    };
    auto fold = [&](float* sc, float* sh, int bias_i, int bn_i, int C) {
        fold_bn_kernel<<<ceil_div(C, 256), 256, 0, stream>>>(
            sc, sh, (const float*)d_in[bias_i],
            (const float*)d_in[bn_i], (const float*)d_in[bn_i + 1],
            (const float*)d_in[bn_i + 2], (const float*)d_in[bn_i + 3], C);
    };
    auto prepge = [&](float* GE, float* GN, int mi, int nf) {
        prep_ge_kernel<<<ceil_div(nf * 3, 256), 256, 0, stream>>>(
            GE, GN, (const float*)d_in[mi + 3], (const float*)d_in[mi + 4],
            (const float*)d_in[mi + 5], nf);
    };
    auto feat = [&](const float* X, float* Fo, int mi, int C, int Vin, int nvp,
                    const float* GE, const float* GN) {
        int total = B * C * nvp;
        build_feat_kernel<<<ceil_div(total, 256), 256, 0, stream>>>(
            X, Fo, (const int*)d_in[mi + 0], (const float*)d_in[mi + 1],
            (const int*)d_in[mi + 2], GE, GN,
            (const int*)d_in[mi + 6], (const float*)d_in[mi + 7],
            total, Vin, nvp);
    };
    auto gemm = [&](const _Float16* Wm, const float* X, const float* sc, const float* sh,
                    const float* res, float* out, int M, int N, int K, int ldx, int flags) {
        dim3 g(ceil_div(N, BN), ceil_div(M, BM), B);
        gemm_bn_kernel<<<g, 256, 0, stream>>>(Wm, X, sc, sh, res, out, M, N, K, ldx, flags);
    };

    // -------- prep --------
    cvt(IN_COEFFS,   Wc0, 32 * 24);
    cvt(B1_BASE + 0, W11, 32 * 32);
    cvt(B1_BASE + 2, W12, 32 * 128);
    cvt(B1_BASE + 4, W13, 128 * 32);
    cvt(B1_BASE + 6, W1S, 128 * 32);
    cvt(B2_BASE + 0, W21, 128 * 128);
    cvt(B2_BASE + 2, W22, 128 * 512);
    cvt(B2_BASE + 4, W23, 512 * 128);
    cvt(B2_BASE + 6, W2S, 512 * 128);
    cvt(B3_BASE + 0, W31, 512 * 512);
    cvt(B3_BASE + 2, W32, 512 * 2048);
    cvt(B3_BASE + 4, W33, 2048 * 512);
    cvt(B3_BASE + 6, W3S, 2048 * 512);

    prepge(GE5, GN5, M5_BASE, NF5);
    prepge(GE3, GN3, M3_BASE, NF3);
    prepge(GE2, GN2, M2_BASE, NF2);
    prepge(GE1, GN1, M1_BASE, NF1);

    fold(sc_in, sh_in, IN_BIAS, IN_BN, 32);
    fold(sc11, sh11, B1_BASE + 1, B1_BASE + 8,  32);
    fold(sc12, sh12, B1_BASE + 3, B1_BASE + 12, 32);
    fold(sc13, sh13, B1_BASE + 5, B1_BASE + 16, 128);
    fold(sc1s, sh1s, B1_BASE + 7, B1_BASE + 20, 128);
    fold(sc21, sh21, B2_BASE + 1, B2_BASE + 8,  128);
    fold(sc22, sh22, B2_BASE + 3, B2_BASE + 12, 128);
    fold(sc23, sh23, B2_BASE + 5, B2_BASE + 16, 512);
    fold(sc2s, sh2s, B2_BASE + 7, B2_BASE + 20, 512);
    fold(sc31, sh31, B3_BASE + 1, B3_BASE + 8,  512);
    fold(sc32, sh32, B3_BASE + 3, B3_BASE + 12, 512);
    fold(sc33, sh33, B3_BASE + 5, B3_BASE + 16, 2048);
    fold(sc3s, sh3s, B3_BASE + 7, B3_BASE + 20, 2048);

    // -------- input mesh conv on m5 (stride 2): (64,6,10242) -> (64,32,2562) --------
    feat((const float*)d_in[IN_X], F, M5_BASE, 6, NV5, NVP5, GE5, GN5);
    gemm(Wc0, F, sc_in, sh_in, nullptr, P0, 32, NVP5, 24, NVP5, FLAG_RELU);

    // -------- block 1 (mesh m3): (64,32,2562) -> (64,128,642) --------
    gemm(W11, P0, sc11, sh11, nullptr, H1, 32, NV3, 32, NVP5, FLAG_RELU);
    feat(H1, F, M3_BASE, 32, NV3, NV3, GE3, GN3);
    gemm(W12, F,  sc12, sh12, nullptr, H2, 32, NV3, 128, NV3, FLAG_RELU);
    gemm(W13, H2, sc13, sh13, nullptr, F, 128, NV3, 32, NV3, 0);
    gemm(W1S, P0, sc1s, sh1s, F, P1, 128, NV3, 32, NVP5, FLAG_RELU | FLAG_ADDRES);

    // -------- block 2 (mesh m2): (64,128,642) -> (64,512,162) --------
    gemm(W21, P1, sc21, sh21, nullptr, H1, 128, NV2, 128, NV3, FLAG_RELU);
    feat(H1, F, M2_BASE, 128, NV2, NV2, GE2, GN2);
    gemm(W22, F,  sc22, sh22, nullptr, H2, 128, NV2, 512, NV2, FLAG_RELU);
    gemm(W23, H2, sc23, sh23, nullptr, F, 512, NV2, 128, NV2, 0);
    gemm(W2S, P1, sc2s, sh2s, F, P0, 512, NV2, 128, NV3, FLAG_RELU | FLAG_ADDRES);

    // -------- block 3 (mesh m1): (64,512,162) -> (64,2048,42) --------
    gemm(W31, P0, sc31, sh31, nullptr, H1, 512, NV1, 512, NV2, FLAG_RELU);
    feat(H1, F, M1_BASE, 512, NV1, NV1, GE1, GN1);
    gemm(W32, F,  sc32, sh32, nullptr, H2, 512, NV1, 2048, NV1, FLAG_RELU);
    gemm(W33, H2, sc33, sh33, nullptr, F, 2048, NV1, 512, NV1, 0);
    gemm(W3S, P0, sc3s, sh3s, F, P1, 2048, NV1, 512, NV2, FLAG_RELU | FLAG_ADDRES);

    // -------- mean pool + FC + log_softmax --------
    mean_pool_kernel<<<ceil_div(B * 2048, 256), 256, 0, stream>>>(P1, pooled, B * 2048, NV1);
    fc_lsm_kernel<<<B, 64, 0, stream>>>(pooled, (const float*)d_in[IN_FCW],
                                        (const float*)d_in[IN_FCB], (float*)d_out);
}
